// Encoder_46943992545741
// MI455X (gfx1250) — compile-verified, matrix operand order checked by reading, not run on hardware
//
#include <hip/hip_runtime.h>

// ---------------------------------------------------------------------------
// Problem constants (match reference): NU=NV=4096, D=256, H=64, R=5, B=4096
// ---------------------------------------------------------------------------
#define NUc 4096
#define NVc 4096
#define Dc  256
#define Hc  64
#define Rc  5
#define Bc  4096

typedef __attribute__((ext_vector_type(16))) __bf16 v16bf;
typedef __attribute__((ext_vector_type(8)))  float  v8f;

// LDS row stride (dwords) for the 32x128 transpose tile: 132 keeps float4
// store alignment and makes transposed reads conflict-free (bank = (4k+c)%64;
// lanes 0-15 and 16-31 land in disjoint bank ranges).
#define TSTRIDE 132

// ---------------------------------------------------------------------------
// Fragment packers (CDNA5 WMMA 16x16x32 bf16, ISA 7.12.2 layouts)
// A (16x32 row-major): lane L holds row M=L%16; kb=(L/16)*8:
//   elements 0-7 = K {kb..kb+7}, elements 8-15 = K {16+kb..16+kb+7}.
// ---------------------------------------------------------------------------
__device__ __forceinline__ v16bf pack_a_rowmajor(const float* __restrict__ p) {
  float4 fa = *(const float4*)(p);
  float4 fb = *(const float4*)(p + 4);
  float4 fc = *(const float4*)(p + 16);
  float4 fd = *(const float4*)(p + 20);
  v16bf a;
  a[0]  = (__bf16)fa.x; a[1]  = (__bf16)fa.y; a[2]  = (__bf16)fa.z; a[3]  = (__bf16)fa.w;
  a[4]  = (__bf16)fb.x; a[5]  = (__bf16)fb.y; a[6]  = (__bf16)fb.z; a[7]  = (__bf16)fb.w;
  a[8]  = (__bf16)fc.x; a[9]  = (__bf16)fc.y; a[10] = (__bf16)fc.z; a[11] = (__bf16)fc.w;
  a[12] = (__bf16)fd.x; a[13] = (__bf16)fd.y; a[14] = (__bf16)fd.z; a[15] = (__bf16)fd.w;
  return a;
}

// Strided 8+8 K pattern (LDS transpose reads with stride TSTRIDE).
__device__ __forceinline__ v16bf pack_strided(const float* p, int stride) {
  v16bf b;
#pragma unroll
  for (int j = 0; j < 8; ++j) {
    b[j]     = (__bf16)p[(size_t)j * stride];
    b[j + 8] = (__bf16)p[(size_t)(16 + j) * stride];
  }
  return b;
}

// ---------------------------------------------------------------------------
// K1: row sums of support  (sum over NV)  -> row_sum[R*NU]; wave per row
// ---------------------------------------------------------------------------
__global__ void k_rowsum(const float* __restrict__ supp, float* __restrict__ row_sum) {
  int wave = threadIdx.x >> 5, lane = threadIdx.x & 31;
  int row = blockIdx.x * 8 + wave;                 // [0, R*NU)
  const float* p = supp + (size_t)row * NVc;
  float s = 0.f;
  for (int i = lane * 4; i < NVc; i += 128) {
    float4 f = *(const float4*)(p + i);
    s += f.x + f.y + f.z + f.w;
  }
#pragma unroll
  for (int off = 16; off > 0; off >>= 1) s += __shfl_xor(s, off, 32);
  if (lane == 0) row_sum[row] = s;
}

// ---------------------------------------------------------------------------
// K2: column sums (sum over NU) -> col_sum[R*NV]; thread per column, coalesced
// ---------------------------------------------------------------------------
__global__ void k_colsum(const float* __restrict__ supp, float* __restrict__ col_sum) {
  int r = blockIdx.x >> 4;
  int c = ((blockIdx.x & 15) << 8) + threadIdx.x;
  const float* p = supp + (size_t)r * NUc * NVc + c;
  float s = 0.f;
  for (int n = 0; n < NUc; ++n) s += p[(size_t)n * NVc];
  col_sum[r * NVc + c] = s;
}

// ---------------------------------------------------------------------------
// K3: guarded rsqrt of both sums
// ---------------------------------------------------------------------------
__global__ void k_inv(const float* __restrict__ row_sum, const float* __restrict__ col_sum,
                      float* __restrict__ c_inv, float* __restrict__ r_inv) {
  int i = blockIdx.x * blockDim.x + threadIdx.x;   // [0, R*NU)
  float rs = row_sum[i]; c_inv[i] = rs > 0.f ? rsqrtf(rs) : 0.f;
  float cs = col_sum[i]; r_inv[i] = cs > 0.f ? rsqrtf(cs) : 0.f;
}

// ---------------------------------------------------------------------------
// K4: cumsum over relations of both weight tensors [R,D,H]
// ---------------------------------------------------------------------------
__global__ void k_cumsum(const float* __restrict__ uwt, const float* __restrict__ vwt,
                         float* __restrict__ uw, float* __restrict__ vw) {
  int i = blockIdx.x * blockDim.x + threadIdx.x;   // [0, D*H)
  float su = 0.f, sv = 0.f;
#pragma unroll
  for (int r = 0; r < Rc; ++r) {
    su += uwt[r * Dc * Hc + i]; uw[r * Dc * Hc + i] = su;
    sv += vwt[r * Dc * Hc + i]; vw[r * Dc * Hc + i] = sv;
  }
}

// ---------------------------------------------------------------------------
// K5: per-(r,b) outer scales: cu[r,b]=c_inv[r,u[b]], rv[r,b]=r_inv[r,v[b]]
// ---------------------------------------------------------------------------
__global__ void k_gscale(const int* __restrict__ u, const int* __restrict__ v,
                         const float* __restrict__ c_inv, const float* __restrict__ r_inv,
                         float* __restrict__ cu, float* __restrict__ rv) {
  int i = blockIdx.x * blockDim.x + threadIdx.x;   // [0, R*B)
  int r = i / Bc, b = i - r * Bc;
  cu[i] = c_inv[r * NUc + u[b]];
  rv[i] = r_inv[r * NVc + v[b]];
}

// ---------------------------------------------------------------------------
// K6: repack cumsum'd weights [R,D,H] f32 into WMMA B fragments (bf16):
// frag id = ((r*8 + kt)*4 + nt), 32 lanes x 16 bf16; lane=(khalf,n):
//   element e -> K = kt*32 + khalf*8 + (e&7) + (e>=8 ? 16 : 0), N = nt*16+n.
// ---------------------------------------------------------------------------
__global__ void k_wrepack(const float* __restrict__ uw, const float* __restrict__ vw,
                          __bf16* __restrict__ uw_frag, __bf16* __restrict__ vw_frag) {
  int t = blockIdx.x * blockDim.x + threadIdx.x;   // [0, 2*R*8*4*32)
  int lane = t & 31;
  int nt   = (t >> 5) & 3;
  int kt   = (t >> 7) & 7;
  int r    = (t >> 10) % Rc;
  int side = t / (Rc << 10);
  const float* src = side ? vw : uw;
  __bf16* dst      = side ? vw_frag : uw_frag;
  int khalf = lane >> 4, n = lane & 15;
  size_t base = (((size_t)(r * 8 + kt) * 4 + nt) * 32 + lane) * 16;
#pragma unroll
  for (int e = 0; e < 16; ++e) {
    int k = kt * 32 + khalf * 8 + (e & 7) + ((e >> 3) << 4);
    dst[base + e] = (__bf16)src[((size_t)r * Dc + k) * Hc + nt * 16 + n];
  }
}

// ---------------------------------------------------------------------------
// K7: feature GEMMs, WMMA bf16: tmp[r,m,h] = scale[r,m]*(feat[m,:] . W[r,:,h])
// One wave per (side, r, 16-row m-tile); A = b128 row loads, B = fragment
// vector loads; all 4 h-tiles in-wave so A is loaded once per k-step.
// ---------------------------------------------------------------------------
__global__ void k_featgemm(const float* __restrict__ u_feat, const float* __restrict__ v_feat,
                           const __bf16* __restrict__ uw_frag, const __bf16* __restrict__ vw_frag,
                           const float* __restrict__ c_inv, const float* __restrict__ r_inv,
                           __bf16* __restrict__ tu_plain, __bf16* __restrict__ tv_plain) {
  int wid  = blockIdx.x * 8 + (threadIdx.x >> 5);  // [0, 2*R*256)
  int lane = threadIdx.x & 31;
  int mt   = wid & 255;
  int r    = (wid >> 8) % Rc;
  int side = wid / (Rc * 256);
  const float* feat   = side ? v_feat : u_feat;
  const __bf16* wfrag = (side ? vw_frag : uw_frag) + (size_t)r * 8 * 4 * 512
                        + (size_t)lane * 16;
  const float* scale  = (side ? r_inv : c_inv) + r * NUc;
  __bf16* out         = (side ? tv_plain : tu_plain) + (size_t)r * NUc * Hc;

  int m0 = mt * 16;
  int M = lane & 15, khalf = lane >> 4;

  const float* arow = feat + (size_t)(m0 + M) * Dc + khalf * 8;
  v8f acc[4] = {};
#pragma unroll
  for (int kt = 0; kt < 8; ++kt) {
    v16bf a = pack_a_rowmajor(arow + kt * 32);
    const __bf16* fb = wfrag + (size_t)kt * (4 * 512);
#pragma unroll
    for (int nt = 0; nt < 4; ++nt) {
      v16bf b = *(const v16bf*)(fb + nt * 512);
      acc[nt] = __builtin_amdgcn_wmma_f32_16x16x32_bf16(false, a, false, b, (short)0,
                                                        acc[nt], false, false);
    }
  }
#pragma unroll
  for (int nt = 0; nt < 4; ++nt)
#pragma unroll
    for (int i = 0; i < 8; ++i) {
      int row = m0 + i + khalf * 8;      // C/D: M = vgpr + 8*(lane/16)
      out[(size_t)row * Hc + nt * 16 + M] = (__bf16)(acc[nt][i] * scale[row]);
    }
}

// ---------------------------------------------------------------------------
// K8: repack tmp (row-major bf16 [R,4096,64]) into per-lane WMMA B fragments:
// frag id = ((r*128 + kt)*4 + nt), 32 lanes x 16 bf16, one 32B load per lane.
// ---------------------------------------------------------------------------
__global__ void k_repack(const __bf16* __restrict__ tu_plain, const __bf16* __restrict__ tv_plain,
                         __bf16* __restrict__ tu_frag, __bf16* __restrict__ tv_frag) {
  int t = blockIdx.x * blockDim.x + threadIdx.x;   // [0, 2*R*128*4*32)
  int lane = t & 31;
  int nt   = (t >> 5) & 3;
  int kt   = (t >> 7) & 127;
  int r    = (t >> 14) % Rc;
  int side = t / (Rc << 14);
  const __bf16* src = side ? tv_plain : tu_plain;
  __bf16* dst       = side ? tv_frag : tu_frag;
  int khalf = lane >> 4, n = lane & 15;
  size_t base = (((size_t)(r * 128 + kt) * 4 + nt) * 32 + lane) * 16;
#pragma unroll
  for (int e = 0; e < 16; ++e) {
    int k = kt * 32 + khalf * 8 + (e & 7) + ((e >> 3) << 4);
    dst[base + e] = src[((size_t)r * NUc + k) * Hc + nt * 16 + n];
  }
}

// ---------------------------------------------------------------------------
// K9: dense row-side GEMM:  Q[r,n,h] = sum_m supp[r,n,m] * tv[r,m,h]
// A rows contiguous -> global_load_b128 + prefetch; 4 accumulators cover H=64.
// ---------------------------------------------------------------------------
__global__ void k_qdense(const float* __restrict__ supp, const __bf16* __restrict__ tv_frag,
                         float* __restrict__ Q) {
  const int KT = NVc / 32, NT = Hc / 16;
  int wid  = blockIdx.x * 8 + (threadIdx.x >> 5); // [0, R*256)
  int lane = threadIdx.x & 31;
  int mt   = wid & 255;
  int r    = wid >> 8;
  int n0 = mt * 16;
  int M = lane & 15, khalf = lane >> 4;

  const float* arow = supp + ((size_t)r * NUc + n0 + M) * NVc + khalf * 8;
  const __bf16* fragr = tv_frag + (size_t)r * KT * NT * 512 + (size_t)lane * 16;

  v8f acc[4] = {};
  for (int kt = 0; kt < KT; ++kt) {
    v16bf a = pack_a_rowmajor(arow + kt * 32);
    if (kt + 1 < KT) __builtin_prefetch(arow + (kt + 1) * 32, 0, 1);
    const __bf16* fb = fragr + (size_t)kt * (NT * 512);
#pragma unroll
    for (int nt = 0; nt < 4; ++nt) {
      v16bf b = *(const v16bf*)(fb + nt * 512);
      acc[nt] = __builtin_amdgcn_wmma_f32_16x16x32_bf16(false, a, false, b, (short)0,
                                                        acc[nt], false, false);
    }
  }
  float* qr = Q + (size_t)r * NUc * Hc;
#pragma unroll
  for (int nt = 0; nt < 4; ++nt)
#pragma unroll
    for (int i = 0; i < 8; ++i)
      qr[(size_t)(n0 + i + khalf * 8) * Hc + nt * 16 + M] = acc[nt][i];
}

// ---------------------------------------------------------------------------
// K10: dense transposed-side GEMM: P[r,c,h] = sum_n supp[r,n,c] * tu[r,n,h]
// Block = (r, 128-column stripe).  Per 32-row k-step the block cooperatively
// loads the 32x128 f32 tile with coalesced b128 loads into LDS (stride-132
// padding, conflict-free both directions), double-buffered; each of the 8
// waves then packs its transposed A fragment from LDS (ds_load) and runs
// 4 WMMAs.  Support is read exactly once, with b128-sized requests only.
// ---------------------------------------------------------------------------
__global__ void k_pdense(const float* __restrict__ supp, const __bf16* __restrict__ tu_frag,
                         float* __restrict__ P) {
  const int KT = NUc / 32, NT = Hc / 16;
  __shared__ float tile[2][32 * TSTRIDE];          // 2 x 16.5 KB

  int r  = blockIdx.x >> 5;                        // [0, R)
  int ct = blockIdx.x & 31;                        // 128-col stripe
  int cbase = ct * 128;
  int w = threadIdx.x >> 5, lane = threadIdx.x & 31;
  int M = lane & 15, khalf = lane >> 4;
  int c0 = cbase + w * 16;

  // cooperative loader mapping: 32 rows x 8 chunks of 16 floats
  int lrow = threadIdx.x >> 3;
  int lcol = (threadIdx.x & 7) * 16;
  const float* sbase = supp + (size_t)r * NUc * NVc + cbase + lcol + (size_t)lrow * NVc;

  const __bf16* fragr = tu_frag + (size_t)r * KT * NT * 512 + (size_t)lane * 16;

  // preload tile 0
  {
    const float* s = sbase;
    float* d = &tile[0][lrow * TSTRIDE + lcol];
    float4 f0 = *(const float4*)(s);
    float4 f1 = *(const float4*)(s + 4);
    float4 f2 = *(const float4*)(s + 8);
    float4 f3 = *(const float4*)(s + 12);
    *(float4*)(d) = f0; *(float4*)(d + 4) = f1;
    *(float4*)(d + 8) = f2; *(float4*)(d + 12) = f3;
  }
  __syncthreads();

  v8f acc[4] = {};
  for (int kt = 0; kt < KT; ++kt) {
    int buf = kt & 1;
    // prefetch next 32-row tile into the other buffer
    if (kt + 1 < KT) {
      const float* s = sbase + (size_t)(kt + 1) * 32 * NVc;
      float* d = &tile[buf ^ 1][lrow * TSTRIDE + lcol];
      float4 f0 = *(const float4*)(s);
      float4 f1 = *(const float4*)(s + 4);
      float4 f2 = *(const float4*)(s + 8);
      float4 f3 = *(const float4*)(s + 12);
      *(float4*)(d) = f0; *(float4*)(d + 4) = f1;
      *(float4*)(d + 8) = f2; *(float4*)(d + 12) = f3;
    }
    // transposed A fragment from LDS: A[c, k] = tile[k][c_local]
    const float* lp = &tile[buf][khalf * 8 * TSTRIDE + w * 16 + M];
    v16bf a = pack_strided(lp, TSTRIDE);
    const __bf16* fb = fragr + (size_t)kt * (NT * 512);
#pragma unroll
    for (int nt = 0; nt < 4; ++nt) {
      v16bf b = *(const v16bf*)(fb + nt * 512);
      acc[nt] = __builtin_amdgcn_wmma_f32_16x16x32_bf16(false, a, false, b, (short)0,
                                                        acc[nt], false, false);
    }
    __syncthreads();
  }

  float* pr = P + (size_t)r * NVc * Hc;
#pragma unroll
  for (int nt = 0; nt < 4; ++nt)
#pragma unroll
    for (int i = 0; i < 8; ++i)
      pr[(size_t)(c0 + i + khalf * 8) * Hc + nt * 16 + M] = acc[nt][i];
}

// ---------------------------------------------------------------------------
// K11: gather + relation-sum + bias + relu:
//   z_u[b,h] = relu(bias[h] + sum_r cu[r,b]*Q[r,u[b],h])
//   z_v[b,h] = relu(bias[h] + sum_r rv[r,b]*P[r,v[b],h])
// ---------------------------------------------------------------------------
__global__ void k_combine(const int* __restrict__ u, const int* __restrict__ v,
                          const float* __restrict__ cu, const float* __restrict__ rv,
                          const float* __restrict__ Q, const float* __restrict__ P,
                          const float* __restrict__ u_bias, float* __restrict__ out) {
  int i = blockIdx.x * blockDim.x + threadIdx.x;   // [0, B*H)
  int b = i >> 6, h = i & 63;
  int urow = u[b], vcol = v[b];
  float zu = 0.f, zv = 0.f;
#pragma unroll
  for (int r = 0; r < Rc; ++r) {
    zu += cu[r * Bc + b] * Q[((size_t)r * NUc + urow) * Hc + h];
    zv += rv[r * Bc + b] * P[((size_t)r * NVc + vcol) * Hc + h];
  }
  float bias = u_bias[h];
  float a = zu + bias, c = zv + bias;
  out[i] = a > 0.f ? a : 0.f;
  out[(size_t)Bc * Hc + i] = c > 0.f ? c : 0.f;
}

// ---------------------------------------------------------------------------
// Workspace layout (bytes):
//   0        row_sum   81920       245760   r_inv    81920
//   81920    col_sum   81920       327680   uw       327680
//   163840   c_inv     81920       655360   vw       327680
//   983040   tu_plain  2621440     3604480  tv_plain 2621440
//   6225920  tu_frag   2621440     8847360  tv_frag  2621440
//   11468800 cu        81920       11550720 rv       81920
//   11632640 Q         5242880     16875520 P        5242880
//   22118400 uw_frag   163840      22282240 vw_frag  163840   total ~22.4 MB
// ---------------------------------------------------------------------------
extern "C" void kernel_launch(void* const* d_in, const int* in_sizes, int n_in,
                              void* d_out, int out_size, void* d_ws, size_t ws_size,
                              hipStream_t stream) {
  const float* u_feat   = (const float*)d_in[0];
  const float* v_feat   = (const float*)d_in[1];
  const int*   u        = (const int*)d_in[2];
  const int*   v        = (const int*)d_in[3];
  const float* supp     = (const float*)d_in[4];
  const float* u_weight = (const float*)d_in[5];
  const float* v_weight = (const float*)d_in[6];
  const float* u_bias   = (const float*)d_in[7];
  float* out = (float*)d_out;

  char* ws = (char*)d_ws;
  float*  row_sum  = (float*)(ws + 0);
  float*  col_sum  = (float*)(ws + 81920);
  float*  c_inv    = (float*)(ws + 163840);
  float*  r_inv    = (float*)(ws + 245760);
  float*  uw       = (float*)(ws + 327680);
  float*  vw       = (float*)(ws + 655360);
  __bf16* tu_plain = (__bf16*)(ws + 983040);
  __bf16* tv_plain = (__bf16*)(ws + 3604480);
  __bf16* tu_frag  = (__bf16*)(ws + 6225920);
  __bf16* tv_frag  = (__bf16*)(ws + 8847360);
  float*  cu       = (float*)(ws + 11468800);
  float*  rv       = (float*)(ws + 11550720);
  float*  Q        = (float*)(ws + 11632640);
  float*  P        = (float*)(ws + 16875520);
  __bf16* uw_frag  = (__bf16*)(ws + 22118400);
  __bf16* vw_frag  = (__bf16*)(ws + 22282240);

  k_rowsum  <<<2560, 256, 0, stream>>>(supp, row_sum);
  k_colsum  <<<80,   256, 0, stream>>>(supp, col_sum);
  k_inv     <<<80,   256, 0, stream>>>(row_sum, col_sum, c_inv, r_inv);
  k_cumsum  <<<64,   256, 0, stream>>>(u_weight, v_weight, uw, vw);
  k_gscale  <<<80,   256, 0, stream>>>(u, v, c_inv, r_inv, cu, rv);
  k_wrepack <<<40,   256, 0, stream>>>(uw, vw, uw_frag, vw_frag);
  k_featgemm<<<320,  256, 0, stream>>>(u_feat, v_feat, uw_frag, vw_frag, c_inv, r_inv,
                                       tu_plain, tv_plain);
  k_repack  <<<640,  256, 0, stream>>>(tu_plain, tv_plain, tu_frag, tv_frag);
  k_qdense  <<<160,  256, 0, stream>>>(supp, tv_frag, Q);
  k_pdense  <<<160,  256, 0, stream>>>(supp, tu_frag, P);
  k_combine <<<1024, 256, 0, stream>>>(u, v, cu, rv, Q, P, u_bias, out);
}
